// FFSpikingLayer_62723702391147
// MI455X (gfx1250) — matrix operand order repaired and checked by reading, not compiled, not run on hardware
//
#include <hip/hip_runtime.h>

// ---- CDNA5 async global->LDS path (guarded: falls back to sync staging) ----
#if defined(__has_builtin)
#  if __has_builtin(__builtin_amdgcn_global_load_async_to_lds_b128)
#    define USE_ASYNC_LDS 1
#  endif
#  if __has_builtin(__builtin_amdgcn_s_wait_asynccnt)
#    define WAIT_ASYNCCNT0() __builtin_amdgcn_s_wait_asynccnt(0)
#  endif
#endif
#ifndef USE_ASYNC_LDS
#  define USE_ASYNC_LDS 0
#endif
#ifndef WAIT_ASYNCCNT0
#  define WAIT_ASYNCCNT0() asm volatile("s_wait_asynccnt 0x0" ::: "memory")
#endif

typedef __attribute__((ext_vector_type(4)))  float  v4f;
typedef __attribute__((ext_vector_type(8)))  float  v8f;
typedef __attribute__((ext_vector_type(4)))  __bf16 v4bf;
typedef __attribute__((ext_vector_type(8)))  __bf16 v8bf;
typedef __attribute__((ext_vector_type(16))) __bf16 v16bf;

// exact pointee type the async-LDS builtin expects: int __vector(4)
typedef int v4i_ __attribute__((vector_size(16)));
typedef __attribute__((address_space(1))) v4i_ gv4i;   // global (__device__) int4
typedef __attribute__((address_space(3))) v4i_ lv4i;   // LDS (__shared__) int4

typedef __attribute__((address_space(3))) float       lds_f;
typedef __attribute__((address_space(1))) const float gbl_cf;

static constexpr int T_STEPS = 32;
static constexpr int BATCH   = 256;
static constexpr int FEAT    = 2048;
static constexpr int HID     = 4096;
static constexpr int NT      = 128;           // hidden cols per block (8 waves x 16)
static constexpr int APAD    = 8;             // bf16 row padding (bank-conflict avoidance)
static constexpr int AROW    = FEAT + APAD;   // 2056 bf16 per staged row

#if USE_ASYNC_LDS
__global__ __launch_bounds__(256, 1)
#else
__global__ __launch_bounds__(256, 2)
#endif
void ff_lif_wmma_kernel(const float* __restrict__ X,   // [T,B,F]
                        const float* __restrict__ Wm,  // [H,F]
                        const float* __restrict__ bias,// [H]
                        float* __restrict__ out)       // [T,B,H] spikes ++ [B,H] count
{
#if USE_ASYNC_LDS
    __shared__ float sRaw[16 * FEAT];        // 128 KB raw fp32 tile (async landing zone)
#endif
    __shared__ __bf16 sAh[16 * AROW];        // bf16 hi
    __shared__ __bf16 sAl[16 * AROW];        // bf16 lo (residual)

    const int tid  = threadIdx.x;
    const int lane = tid & 31;
    const int wave = tid >> 5;
    const int col  = lane & 15;     // N within wave's 16x16 tile; also A-row index
    const int half = lane >> 4;     // which half-wave (selects K sub-runs)
    const int n0   = blockIdx.x * NT + wave * 16;
    const int b0   = blockIdx.y * 16;

    const float* wrow = Wm + (size_t)(n0 + col) * FEAT;  // B operand: W[h, :] (K-contig)
    const float  bval = bias[n0 + col];

    v8f v   = {};   // membrane potential for this lane's 8 (m,n) cells
    v8f cnt = {};   // spike count accumulator

#if USE_ASYNC_LDS
    lds_f* raw3 = (lds_f*)sRaw;
    // prologue: async-copy tile t=0 into sRaw (32 x b128 per thread)
    {
        gbl_cf* xg = (gbl_cf*)(X + (size_t)b0 * FEAT);
        #pragma unroll 4
        for (int j = 0; j < 32; ++j) {
            int f4 = tid + j * 256;
            __builtin_amdgcn_global_load_async_to_lds_b128(
                (gv4i*)(xg + (size_t)f4 * 4),
                (lv4i*)(raw3 + (size_t)f4 * 4), 0, 0);
        }
    }
#endif

    for (int t = 0; t < T_STEPS; ++t) {
#if USE_ASYNC_LDS
        WAIT_ASYNCCNT0();              // this wave's async copies into sRaw done
#endif
        __syncthreads();               // tile ready (async) / prev hi-lo consumed

        // ---- convert A(t) tile [16 x FEAT] to bf16 hi/lo in LDS ----
        const float* xt = X + (size_t)t * (BATCH * FEAT) + (size_t)b0 * FEAT;
        #pragma unroll 4
        for (int j = 0; j < 32; ++j) {
            int f4 = tid + j * 256;        // float4 index 0..8191 over the 16x2048 tile
            int r  = f4 >> 9;              // row 0..15
            int kc = (f4 & 511) << 2;      // float offset in row
#if USE_ASYNC_LDS
            v4f q = *(const v4f*)(sRaw + (size_t)f4 * 4);
#else
            v4f q = *(const v4f*)(xt + (size_t)r * FEAT + kc);
#endif
            v4bf h, l;
            #pragma unroll
            for (int i = 0; i < 4; ++i) {
                __bf16 hb = (__bf16)q[i];
                h[i] = hb;
                l[i] = (__bf16)(q[i] - (float)hb);
            }
            *(v4bf*)(sAh + r * AROW + kc) = h;
            *(v4bf*)(sAl + r * AROW + kc) = l;
        }
        __syncthreads();               // hi/lo ready; sRaw free for next tile

#if USE_ASYNC_LDS
        // ---- overlap: async-copy tile t+1 while computing tile t ----
        if (t + 1 < T_STEPS) {
            gbl_cf* xg = (gbl_cf*)(X + (size_t)(t + 1) * (BATCH * FEAT)
                                     + (size_t)b0 * FEAT);
            #pragma unroll 4
            for (int j = 0; j < 32; ++j) {
                int f4 = tid + j * 256;
                __builtin_amdgcn_global_load_async_to_lds_b128(
                    (gv4i*)(xg + (size_t)f4 * 4),
                    (lv4i*)(raw3 + (size_t)f4 * 4), 0, 0);
            }
        }
#endif

        v8f acc = {};
        // ---- 16x16 tile GEMM over K=2048, bf16x2 split (hi*hi + hi*lo + lo*hi) ----
        for (int ks = 0; ks < FEAT; ks += 32) {
            // B operand: runs [ks+half*8, +8) and [ks+half*8+16, +8) (matches A)
            const float* wp = wrow + ks + half * 8;
            __builtin_prefetch(wp + 64, 0, 0);  // 2 K-steps ahead -> global_prefetch_b8
            v4f w0 = *(const v4f*)(wp);
            v4f w1 = *(const v4f*)(wp + 4);
            v4f w2 = *(const v4f*)(wp + 16);
            v4f w3 = *(const v4f*)(wp + 20);
            v16bf bh, bl;
            #pragma unroll
            for (int i = 0; i < 16; ++i) {
                float f = (i < 4) ? w0[i] : (i < 8) ? w1[i - 4]
                        : (i < 12) ? w2[i - 8] : w3[i - 12];
                __bf16 hb = (__bf16)f;
                bh[i] = hb;
                bl[i] = (__bf16)(f - (float)hb);
            }
            // A operand from LDS: runs [ks+half*8, +8) and [ks+half*8+16, +8)
            const __bf16* ahp = sAh + col * AROW + ks + half * 8;
            const __bf16* alp = sAl + col * AROW + ks + half * 8;
            v8bf ah0 = *(const v8bf*)(ahp);
            v8bf ah1 = *(const v8bf*)(ahp + 16);
            v8bf al0 = *(const v8bf*)(alp);
            v8bf al1 = *(const v8bf*)(alp + 16);
            v16bf ah = __builtin_shufflevector(ah0, ah1, 0,1,2,3,4,5,6,7,8,9,10,11,12,13,14,15);
            v16bf al = __builtin_shufflevector(al0, al1, 0,1,2,3,4,5,6,7,8,9,10,11,12,13,14,15);

            acc = __builtin_amdgcn_wmma_f32_16x16x32_bf16(false, ah, false, bh,
                                                          (short)0, acc, false, false);
            acc = __builtin_amdgcn_wmma_f32_16x16x32_bf16(false, ah, false, bl,
                                                          (short)0, acc, false, false);
            acc = __builtin_amdgcn_wmma_f32_16x16x32_bf16(false, al, false, bh,
                                                          (short)0, acc, false, false);
        }

        // ---- fused LIF: v=(v+cur)/2, spike=(v>=1), hard reset, store spikes ----
        float* orow = out + ((size_t)t * BATCH + b0) * HID + (size_t)(n0 + col);
        #pragma unroll
        for (int r = 0; r < 8; ++r) {
            float cur = acc[r] + bval;
            float vm  = 0.5f * (v[r] + cur);
            bool  s   = (vm >= 1.0f);
            float spk = s ? 1.0f : 0.0f;
            v[r]   = s ? 0.0f : vm;
            cnt[r] += spk;
            int m = r + half * 8;       // C/D layout: VGPR r -> row m (lanes 16-31: +8)
            orow[(size_t)m * HID] = spk;
        }
        (void)xt;
    }

    // ---- spike counts: out[T*B*H + b*H + h] ----
    float* crow = out + (size_t)T_STEPS * BATCH * HID
                      + (size_t)b0 * HID + (size_t)(n0 + col);
    #pragma unroll
    for (int r = 0; r < 8; ++r) {
        int m = r + half * 8;
        crow[(size_t)m * HID] = cnt[r];
    }
}

extern "C" void kernel_launch(void* const* d_in, const int* in_sizes, int n_in,
                              void* d_out, int out_size, void* d_ws, size_t ws_size,
                              hipStream_t stream) {
    (void)in_sizes; (void)n_in; (void)out_size; (void)d_ws; (void)ws_size;
    const float* X  = (const float*)d_in[0];  // [32,256,2048]
    const float* Wm = (const float*)d_in[1];  // [4096,2048]
    const float* b  = (const float*)d_in[2];  // [4096]
    float* out = (float*)d_out;               // 32*256*4096 + 256*4096 floats

    dim3 grid(HID / NT, BATCH / 16);          // 32 x 16 blocks
    dim3 block(256);                          // 8 wave32s
    ff_lif_wmma_kernel<<<grid, block, 0, stream>>>(X, Wm, b, out);
}
